// GDNSequenceModel_56693568307361
// MI455X (gfx1250) — matrix operand order repaired and checked by reading, not compiled, not run on hardware
//
#include <hip/hip_runtime.h>
#include <cstdint>

// ---------------------------------------------------------------------------
// Gated DeltaNet forward on gfx1250 (wave32, WMMA).
// B=2, T=2048, D=1024, H=8, K=128, V=256, CONV=4
// ---------------------------------------------------------------------------

typedef __attribute__((ext_vector_type(16))) _Float16 v16h;
typedef __attribute__((ext_vector_type(8)))  float    v8f;
typedef __attribute__((ext_vector_type(4)))  float    v4f;
typedef __attribute__((ext_vector_type(4)))  unsigned int v4u;

union FragCvt {
    struct { v4u lo; v4u hi; } p;
    v16h h;
};

// ---------------------------------------------------------------------------
// f32 -> f16 elementwise convert
// ---------------------------------------------------------------------------
__global__ void __launch_bounds__(256)
f32_to_f16_kernel(const float* __restrict__ in, _Float16* __restrict__ out, int n)
{
    int i = blockIdx.x * 256 + threadIdx.x;
    if (i < n) out[i] = (_Float16)in[i];
}

// W: [Kd, N] f32 row-major  ->  Wt: [N, Kd] f16 row-major
__global__ void __launch_bounds__(256)
transpose_f32_to_f16_kernel(const float* __restrict__ W, _Float16* __restrict__ Wt,
                            int Kd, int N)
{
    int i = blockIdx.x * 256 + threadIdx.x;
    if (i < Kd * N) {
        int k = i / N, n = i % N;
        Wt[(size_t)n * Kd + k] = (_Float16)W[i];
    }
}

// ---------------------------------------------------------------------------
// WMMA GEMM: C[M,N] = A[M,Kd] * B[Kd,N], B given transposed (Bt[N,Kd]).
// Block: 256 threads = 8 waves arranged 2(M) x 4(N).
// Wave tile: 32(M) x 64(N) = 2x4 WMMA accumulators -> 8 WMMAs per 12 b128
// loads per k-step (vs 8 loads / 2 WMMAs before).
// Block tile: 64 x 256. Fragment layouts per CDNA5 ISA 7.12.2.
// ---------------------------------------------------------------------------
__global__ void __launch_bounds__(256)
gemm_f16_wmma_kernel(const _Float16* __restrict__ A,   // [M, Kd]
                     const _Float16* __restrict__ Bt,  // [N, Kd]
                     float* __restrict__ C,            // [M, N]
                     int M, int N, int Kd)
{
    const int lane = threadIdx.x & 31;
    const int wave = threadIdx.x >> 5;
    const int wm = wave >> 2;                 // 0..1
    const int wn = wave & 3;                  // 0..3
    const int m0 = blockIdx.x * 64 + wm * 32;
    const int n0 = blockIdx.y * 256 + wn * 64;
    if (m0 >= M || n0 >= N) return;           // uniform per wave

    const int mr   = lane & 15;               // row/col within 16-wide tile
    const int half = lane >> 4;               // lane group 0/1

    // A fragment lane ptrs (rows m0+mr, m0+16+mr):
    //   k = half*8 + {0..7} (VGPR0-3) and 16 + half*8 + {0..7} (VGPR4-7)
    const _Float16* pa0 = A + (size_t)(m0 + mr) * Kd + half * 8;
    const _Float16* pa1 = pa0 + (size_t)16 * Kd;
    // B fragment lane ptrs (cols n0 + 16j + mr): k = half*8 + {0..7} and 16 + half*8 + {0..7}
    const _Float16* pb0 = Bt + (size_t)(n0 + mr) * Kd + half * 8;

    v8f acc[2][4] = {};
    for (int k0 = 0; k0 < Kd; k0 += 32) {
        FragCvt fa[2], fb[4];
        fa[0].p.lo = *(const v4u*)(pa0 + k0);
        fa[0].p.hi = *(const v4u*)(pa0 + k0 + 16);
        fa[1].p.lo = *(const v4u*)(pa1 + k0);
        fa[1].p.hi = *(const v4u*)(pa1 + k0 + 16);
#pragma unroll
        for (int j = 0; j < 4; ++j) {
            const _Float16* pb = pb0 + (size_t)(16 * j) * Kd;
            fb[j].p.lo = *(const v4u*)(pb + k0);
            fb[j].p.hi = *(const v4u*)(pb + k0 + 16);
        }
        // prefetch next k-tiles into cache while WMMAs drain (global_prefetch_b8)
        if (k0 + 64 < Kd) {
            __builtin_prefetch(pa0 + k0 + 64, 0, 1);
            __builtin_prefetch(pb0 + k0 + 64, 0, 1);
        }
#pragma unroll
        for (int i = 0; i < 2; ++i)
#pragma unroll
            for (int j = 0; j < 4; ++j)
                acc[i][j] = __builtin_amdgcn_wmma_f32_16x16x32_f16(
                    false, fa[i].h, false, fb[j].h, (short)0, acc[i][j], false, false);
    }

    // C/D layout: VGPR r -> row +r+8*half, col +mr
#pragma unroll
    for (int i = 0; i < 2; ++i) {
#pragma unroll
        for (int j = 0; j < 4; ++j) {
            float* pc = C + (size_t)(m0 + i * 16 + half * 8) * N + n0 + j * 16 + mr;
#pragma unroll
            for (int r = 0; r < 8; ++r) pc[(size_t)r * N] = acc[i][j][r];
        }
    }
}

// ---------------------------------------------------------------------------
// beta = sigmoid(x@W_b), g = -exp(A_log)*softplus(x@W_a + dt_bias)
// One block per row r in [0, B*T); 16 dots of length D per row.
// ---------------------------------------------------------------------------
__global__ void __launch_bounds__(256)
beta_g_kernel(const float* __restrict__ x, const float* __restrict__ Wb,
              const float* __restrict__ Wa, const float* __restrict__ dtb,
              const float* __restrict__ Alog, float* __restrict__ beta,
              float* __restrict__ g, int Dd)
{
    __shared__ float red[256];
    const int r    = blockIdx.x;
    const int out  = threadIdx.x >> 4;   // 0..15 (0..7 -> beta_h, 8..15 -> g_h)
    const int part = threadIdx.x & 15;
    const int h    = out & 7;
    const float* W = (out < 8) ? Wb : Wa;
    const float* xr = x + (size_t)r * Dd;
    float acc = 0.f;
    for (int i = part; i < Dd; i += 16) acc += xr[i] * W[(size_t)i * 8 + h];
    red[threadIdx.x] = acc;
    __syncthreads();
    if (part == 0) {
        float s = 0.f;
#pragma unroll
        for (int j = 0; j < 16; ++j) s += red[(out << 4) + j];
        if (out < 8) {
            beta[(size_t)r * 8 + h] = 1.f / (1.f + __expf(-s));
        } else {
            float z  = s + dtb[h];
            float sp = (z > 20.f) ? z : log1pf(__expf(z));
            g[(size_t)r * 8 + h] = -__expf(Alog[h]) * sp;
        }
    }
}

// ---------------------------------------------------------------------------
// Causal depthwise conv (CONV=4) + SiLU + per-head L2 norm (+ optional scale).
// Grid: (B*T, H), block 128 threads (one per channel-in-head), C = H*128.
// ---------------------------------------------------------------------------
__global__ void __launch_bounds__(128)
conv_silu_l2_kernel(const float* __restrict__ pre, const float* __restrict__ cw,
                    float* __restrict__ post, float scale, int T, int C)
{
    __shared__ float red[128];
    const int bt = blockIdx.x;
    const int h  = blockIdx.y;
    const int b  = bt / T, t = bt % T;
    const int j  = threadIdx.x;
    const int c  = h * 128 + j;
    float acc = 0.f;
#pragma unroll
    for (int jj = 0; jj < 4; ++jj) {
        int tt = t - 3 + jj;
        if (tt >= 0)
            acc = fmaf(pre[((size_t)(b * T + tt)) * C + c], cw[c * 4 + jj], acc);
    }
    float sv = acc / (1.f + __expf(-acc));          // silu
    red[j] = sv * sv;
    __syncthreads();
    for (int off = 64; off > 0; off >>= 1) {
        if (j < off) red[j] += red[j + off];
        __syncthreads();
    }
    float nrm = fmaxf(sqrtf(red[0]), 1e-6f);
    post[(size_t)bt * C + c] = sv / nrm * scale;
}

// Causal depthwise conv + SiLU (no norm), one thread per (b,t,c).
__global__ void __launch_bounds__(256)
conv_silu_kernel(const float* __restrict__ pre, const float* __restrict__ cw,
                 float* __restrict__ post, int T, int C, int total)
{
    int i = blockIdx.x * 256 + threadIdx.x;
    if (i >= total) return;
    int c  = i % C;
    int bt = i / C;
    int t  = bt % T, b = bt / T;
    float acc = 0.f;
#pragma unroll
    for (int jj = 0; jj < 4; ++jj) {
        int tt = t - 3 + jj;
        if (tt >= 0)
            acc = fmaf(pre[((size_t)(b * T + tt)) * C + c], cw[c * 4 + jj], acc);
    }
    post[i] = acc / (1.f + __expf(-acc));
}

// ---------------------------------------------------------------------------
// Gated delta rule recurrence, barrier-free.
// State S[K=128, V-slice=64] in registers: grid = B*H*(V/64) = 64 blocks,
// 256 threads: thread (vl = tid>>2, kq = tid&3) owns S[kq*32..kq*32+31][vl].
// Each thread loads its own contiguous 32-float k_t/q_t slice directly from
// global (8x b128 each, broadcast across v-threads, L1-served) -> the 2048-
// step serial loop has NO __syncthreads and NO LDS on the critical path.
// k-reduction via wave32 shfl_xor (partners tid^1, tid^2 share vl, same wave).
// ---------------------------------------------------------------------------
__global__ void __launch_bounds__(256)
gdn_recurrence_kernel(const float* __restrict__ q, const float* __restrict__ k,
                      const float* __restrict__ v, const float* __restrict__ beta,
                      const float* __restrict__ g, float* __restrict__ o, int T)
{
    const int bid  = blockIdx.x;        // [0, 64)
    const int vq4  = bid & 3;
    const int h    = (bid >> 2) & 7;
    const int b    = bid >> 5;
    const int vcol = vq4 * 64 + (threadIdx.x >> 2);
    const int kq   = threadIdx.x & 3;

    const size_t rowq = (size_t)h * 128 + kq * 32;

    float s[32];
#pragma unroll
    for (int i = 0; i < 32; ++i) s[i] = 0.f;

    for (int t = 0; t < T; ++t) {
        const size_t r = (size_t)b * T + t;
        const v4f* kp = (const v4f*)(k + r * 1024 + rowq);
        const v4f* qp = (const v4f*)(q + r * 1024 + rowq);
        v4f kk[8], qq[8];
#pragma unroll
        for (int w = 0; w < 8; ++w) { kk[w] = kp[w]; qq[w] = qp[w]; }
        const float* kf = (const float*)kk;
        const float* qf = (const float*)qq;

        const float gt = g[r * 8 + h];
        const float bt = beta[r * 8 + h];
        const float vt = v[r * 2048 + h * 256 + vcol];
        const float eg = __expf(gt);

        float pkv = 0.f;
#pragma unroll
        for (int i = 0; i < 32; ++i) {
            s[i] *= eg;                                   // gated decay
            pkv = fmaf(kf[i], s[i], pkv);                 // kv = k . S
        }
        pkv += __shfl_xor(pkv, 1);
        pkv += __shfl_xor(pkv, 2);

        const float err = (vt - pkv) * bt;                // delta-rule error

        float po = 0.f;
#pragma unroll
        for (int i = 0; i < 32; ++i) {
            s[i] = fmaf(kf[i], err, s[i]);                // S += k (x) err
            po   = fmaf(qf[i], s[i], po);                 // o = q . S
        }
        po += __shfl_xor(po, 1);
        po += __shfl_xor(po, 2);
        if (kq == 0) o[r * 2048 + h * 256 + vcol] = po;
    }
}

// ---------------------------------------------------------------------------
// Gated RMSNorm: on = (o*silu(gate)) * rsqrt(mean((o*silu(gate))^2)+eps)*rms_w
// Block per (r, h): blockIdx.x in [0, B*T*H), 256 threads (one per v).
// Writes f16 for the final WMMA GEMM.
// ---------------------------------------------------------------------------
__global__ void __launch_bounds__(256)
gated_rmsnorm_kernel(const float* __restrict__ o, const float* __restrict__ gate,
                     const float* __restrict__ rms_w, _Float16* __restrict__ onh)
{
    __shared__ float red[256];
    const size_t base = (size_t)blockIdx.x * 256;
    const int vv = threadIdx.x;
    float gt = gate[base + vv];
    float og = o[base + vv] * (gt / (1.f + __expf(-gt)));
    red[vv] = og * og;
    __syncthreads();
    for (int off = 128; off > 0; off >>= 1) {
        if (vv < off) red[vv] += red[vv + off];
        __syncthreads();
    }
    float rs = rsqrtf(red[0] * (1.f / 256.f) + 1e-5f);
    onh[base + vv] = (_Float16)(og * rs * rms_w[vv]);
}

// ---------------------------------------------------------------------------
// Host-side orchestration
// ---------------------------------------------------------------------------
extern "C" void kernel_launch(void* const* d_in, const int* in_sizes, int n_in,
                              void* d_out, int out_size, void* d_ws, size_t ws_size,
                              hipStream_t stream)
{
    (void)in_sizes; (void)n_in; (void)out_size; (void)ws_size;

    const float* x    = (const float*)d_in[0];
    const float* Wq   = (const float*)d_in[1];
    const float* Wk   = (const float*)d_in[2];
    const float* Wv   = (const float*)d_in[3];
    const float* cqw  = (const float*)d_in[4];
    const float* ckw  = (const float*)d_in[5];
    const float* cvw  = (const float*)d_in[6];
    const float* Wb   = (const float*)d_in[7];
    const float* Wa   = (const float*)d_in[8];
    const float* dtb  = (const float*)d_in[9];
    const float* Alog = (const float*)d_in[10];
    const float* Wg   = (const float*)d_in[11];
    const float* rmsw = (const float*)d_in[12];
    const float* Wo   = (const float*)d_in[13];
    float* out = (float*)d_out;

    constexpr int Bb = 2, T = 2048, Dd = 1024, H = 8, Kk = 128, Vv = 256;
    constexpr int M = Bb * T, HK = H * Kk, HV = H * Vv;

    char* wsp = (char*)d_ws;
    size_t off = 0;
    auto alloc = [&](size_t bytes) -> void* {
        void* p = wsp + off;
        off += bytes;
        off = (off + 255) & ~(size_t)255;
        return p;
    };

    _Float16* Xh   = (_Float16*)alloc((size_t)M * Dd * 2);
    _Float16* Wqt  = (_Float16*)alloc((size_t)HK * Dd * 2);
    _Float16* Wkt  = (_Float16*)alloc((size_t)HK * Dd * 2);
    _Float16* Wvt  = (_Float16*)alloc((size_t)HV * Dd * 2);
    _Float16* Wgt  = (_Float16*)alloc((size_t)HV * Dd * 2);
    _Float16* Wot  = (_Float16*)alloc((size_t)Dd * HV * 2);
    float* q_pre   = (float*)alloc((size_t)M * HK * 4);
    float* k_pre   = (float*)alloc((size_t)M * HK * 4);
    float* v_pre   = (float*)alloc((size_t)M * HV * 4);
    float* gatep   = (float*)alloc((size_t)M * HV * 4);
    float* qpost   = (float*)alloc((size_t)M * HK * 4);
    float* kpost   = (float*)alloc((size_t)M * HK * 4);
    float* vpost   = (float*)alloc((size_t)M * HV * 4);
    float* betab   = (float*)alloc((size_t)M * H * 4);
    float* gdec    = (float*)alloc((size_t)M * H * 4);
    float* obuf    = (float*)alloc((size_t)M * HV * 4);
    _Float16* onh  = (_Float16*)alloc((size_t)M * HV * 2);

    // 1) convert x to f16; transpose+convert weights to f16 [N, Kd]
    {
        int n = M * Dd;
        f32_to_f16_kernel<<<(n + 255) / 256, 256, 0, stream>>>(x, Xh, n);
    }
    transpose_f32_to_f16_kernel<<<(Dd * HK + 255) / 256, 256, 0, stream>>>(Wq, Wqt, Dd, HK);
    transpose_f32_to_f16_kernel<<<(Dd * HK + 255) / 256, 256, 0, stream>>>(Wk, Wkt, Dd, HK);
    transpose_f32_to_f16_kernel<<<(Dd * HV + 255) / 256, 256, 0, stream>>>(Wv, Wvt, Dd, HV);
    transpose_f32_to_f16_kernel<<<(Dd * HV + 255) / 256, 256, 0, stream>>>(Wg, Wgt, Dd, HV);
    transpose_f32_to_f16_kernel<<<(HV * Dd + 255) / 256, 256, 0, stream>>>(Wo, Wot, HV, Dd);

    // 2) WMMA projections (block tile 64x256, wave tile 32x64)
    gemm_f16_wmma_kernel<<<dim3(M / 64, HK / 256), 256, 0, stream>>>(Xh, Wqt, q_pre, M, HK, Dd);
    gemm_f16_wmma_kernel<<<dim3(M / 64, HK / 256), 256, 0, stream>>>(Xh, Wkt, k_pre, M, HK, Dd);
    gemm_f16_wmma_kernel<<<dim3(M / 64, HV / 256), 256, 0, stream>>>(Xh, Wvt, v_pre, M, HV, Dd);
    gemm_f16_wmma_kernel<<<dim3(M / 64, HV / 256), 256, 0, stream>>>(Xh, Wgt, gatep, M, HV, Dd);

    // 3) beta / decay gate (small dots, f32 VALU)
    beta_g_kernel<<<M, 256, 0, stream>>>(x, Wb, Wa, dtb, Alog, betab, gdec, Dd);

    // 4) causal conv + SiLU (+ L2 norm for q,k; q also scaled by K^-1/2)
    const float qscale = 1.0f / sqrtf((float)Kk);
    conv_silu_l2_kernel<<<dim3(M, H), 128, 0, stream>>>(q_pre, cqw, qpost, qscale, T, HK);
    conv_silu_l2_kernel<<<dim3(M, H), 128, 0, stream>>>(k_pre, ckw, kpost, 1.0f, T, HK);
    conv_silu_kernel<<<(M * HV + 255) / 256, 256, 0, stream>>>(v_pre, cvw, vpost, T, HV, M * HV);

    // 5) sequential gated delta rule (64 independent v-slices, barrier-free)
    gdn_recurrence_kernel<<<Bb * H * (Vv / 64), 256, 0, stream>>>(
        qpost, kpost, vpost, betab, gdec, obuf, T);

    // 6) gated RMSNorm -> f16
    gated_rmsnorm_kernel<<<M * H, 256, 0, stream>>>(obuf, gatep, rmsw, onh);

    // 7) output projection via WMMA: out[M, D] = onh[M, HV] @ Wo[HV, D]
    gemm_f16_wmma_kernel<<<dim3(M / 64, Dd / 256), 256, 0, stream>>>(onh, Wot, out, M, Dd, HV);
}
